// LSGNN_71511205479062
// MI455X (gfx1250) — matrix operand order repaired and checked by reading, not compiled, not run on hardware
//
#include <hip/hip_runtime.h>

// ---------------------------------------------------------------------------
// LSGNN forward for MI455X (gfx1250, wave32, WMMA + async global->LDS).
// Dense GEMMs (~65 GFLOP) run through v_wmma_f32_16x16x32_bf16 with f32
// accumulation. Interior GEMM tiles stream A/B via double-buffered
// global_load_async_to_lds_b128 (ASYNCcnt), overlapping DMA with WMMA.
// Irregular GCN scatter uses float4 gathers + f32 atomics.
// ---------------------------------------------------------------------------

typedef __bf16 v16bf __attribute__((ext_vector_type(16)));
typedef __bf16 v8bf  __attribute__((ext_vector_type(8)));
typedef float  v8f   __attribute__((ext_vector_type(8)));

#define GG 7
#define NNODE 1024
#define NB 32
#define NROWS (NB * NNODE)   // 32768
#define BM 128
#define BN 128
#define BK 32

__device__ __forceinline__ __bf16 f2bf(float f) {
  unsigned u = __float_as_uint(f);
  unsigned short h = (unsigned short)((u + 0x7FFFu + ((u >> 16) & 1u)) >> 16);
  return __builtin_bit_cast(__bf16, h);
}

// One BK=32 step of the 128x128 tile: 8 x v_wmma_f32_16x16x32_bf16 per wave.
// A frag: lane = M row, contiguous K run half*16..+15 (two b128 LDS reads).
// B frag: lane = N col, contiguous K run half*16..+15 (tile stored N-major).
__device__ __forceinline__ void mma_step(const __bf16* sA, const __bf16* sB,
                                         int wm, int wn, int half, int l16,
                                         v8f (&acc)[2][4]) {
  v16bf af[2];
#pragma unroll
  for (int i = 0; i < 2; ++i) {
    int r = (wm * 32 + i * 16 + l16) * BK;
    v8bf lo = *(const v8bf*)(sA + r + half * 16);
    v8bf hi = *(const v8bf*)(sA + r + half * 16 + 8);
#pragma unroll
    for (int e = 0; e < 8; ++e) { af[i][e] = lo[e]; af[i][8 + e] = hi[e]; }
  }
#pragma unroll
  for (int j = 0; j < 4; ++j) {
    int n = wn * 64 + j * 16 + l16;
    v16bf bfrag = *(const v16bf*)(sB + n * BK + half * 16);
#pragma unroll
    for (int i = 0; i < 2; ++i)
      acc[i][j] = __builtin_amdgcn_wmma_f32_16x16x32_bf16(
          false, af[i], false, bfrag, (short)0, acc[i][j], false, false);
  }
}

// ---------------------------------------------------------------------------
// WMMA GEMM: C[m,n] = act( sum_k A[m,k] * Bt[n,k] + bias[n] )
// A: M x K bf16 row-major.  Bt: Nc x K bf16 (B transposed, K contiguous).
// grid.z batches with per-z element strides sA/sB/sC.
// 256 threads = 8 waves (4x2); each wave owns a 32x64 sub-tile (2x4 WMMAs).
// Interior tiles: double-buffered async global->LDS copies (ASYNCcnt).
// ---------------------------------------------------------------------------
__global__ __launch_bounds__(256) void k_gemm_bf16(
    const __bf16* __restrict__ A, const __bf16* __restrict__ Bt,
    const float* __restrict__ bias, float* __restrict__ Cf,
    __bf16* __restrict__ Cb, int M, int Nc, int K, int act, int ldc,
    long long sA, long long sB, long long sC) {
  __shared__ __align__(64) __bf16 shA[2][BM * BK];
  __shared__ __align__(64) __bf16 shB[2][BN * BK];
  const int z = blockIdx.z;
  A  += (long long)z * sA;
  Bt += (long long)z * sB;
  const long long cofs = (long long)z * sC;
  const int tileM = blockIdx.y * BM;
  const int tileN = blockIdx.x * BN;
  const int t = threadIdx.x;
  const int wave = t >> 5, lane = t & 31;
  const int wm = wave >> 1, wn = wave & 1;   // 4x2 wave grid
  const int half = lane >> 4, l16 = lane & 15;
  const int row0 = t >> 2, kc0 = (t & 3) << 3;   // this thread's 8-elt chunk

  v8f acc[2][4] = {};

  const bool interior =
      (tileM + BM <= M) && (tileN + BN <= Nc) && ((K & (BK - 1)) == 0);

  if (interior) {
    // -------- async double-buffered path (dominant GEMMs) --------
    const unsigned aB[2] = { (unsigned)(unsigned long long)(const void*)&shA[0][0],
                             (unsigned)(unsigned long long)(const void*)&shA[1][0] };
    const unsigned bB[2] = { (unsigned)(unsigned long long)(const void*)&shB[0][0],
                             (unsigned)(unsigned long long)(const void*)&shB[1][0] };
    const __bf16* Abase = A  + (long long)tileM * K;
    const __bf16* Bbase = Bt + (long long)tileN * K;
    // Each wave issues 4 async b128 transfers (16B/lane) per K-tile:
    // 2 for A (rows row0, row0+64) and 2 for B.
    auto issue = [&](int k0, int buf) {
#pragma unroll
      for (int c = 0; c < 2; ++c) {
        int row = row0 + c * 64;
        unsigned la = aB[buf] + (unsigned)((row * BK + kc0) * 2);
        unsigned long long ga =
            (unsigned long long)(Abase + (long long)row * K + k0 + kc0);
        asm volatile("global_load_async_to_lds_b128 %0, %1, off"
                     :: "v"(la), "v"(ga) : "memory");
        unsigned lb = bB[buf] + (unsigned)((row * BK + kc0) * 2);
        unsigned long long gb =
            (unsigned long long)(Bbase + (long long)row * K + k0 + kc0);
        asm volatile("global_load_async_to_lds_b128 %0, %1, off"
                     :: "v"(lb), "v"(gb) : "memory");
      }
    };
    const int nk = K / BK;
    issue(0, 0);
    for (int kt = 0; kt < nk; ++kt) {
      if (kt + 1 < nk) {
        issue((kt + 1) * BK, (kt + 1) & 1);
        // 8 outstanding; retire the older 4 (current tile, in-order).
        asm volatile("s_wait_asynccnt 0x4" ::: "memory");
      } else {
        asm volatile("s_wait_asynccnt 0x0" ::: "memory");
      }
      __syncthreads();                       // all waves' slices landed
      mma_step(shA[kt & 1], shB[kt & 1], wm, wn, half, l16, acc);
      __syncthreads();                       // done reading before overwrite
    }
  } else {
    // -------- guarded synchronous path (boundary tiles) --------
    for (int k0 = 0; k0 < K; k0 += BK) {
#pragma unroll
      for (int c = 0; c < 2; ++c) {
        int row = row0 + c * 64;
        int gk = k0 + kc0;
        int gr = tileM + row;
        v8bf va = {};
        if (gr < M && gk < K) va = *(const v8bf*)(A + (long long)gr * K + gk);
        *(v8bf*)(&shA[0][row * BK + kc0]) = va;
        int gn = tileN + row;
        v8bf vb = {};
        if (gn < Nc && gk < K) vb = *(const v8bf*)(Bt + (long long)gn * K + gk);
        *(v8bf*)(&shB[0][row * BK + kc0]) = vb;
      }
      if (k0 + BK < K)  // global_prefetch_b8 hint for next A tile
        __builtin_prefetch(A + (long long)(tileM + row0) * K + (k0 + BK) + kc0, 0, 1);
      __syncthreads();
      mma_step(shA[0], shB[0], wm, wn, half, l16, acc);
      __syncthreads();
    }
  }

  // Epilogue: C/D layout -> m = v + 8*half, n = l16.
#pragma unroll
  for (int i = 0; i < 2; ++i)
#pragma unroll
    for (int j = 0; j < 4; ++j)
#pragma unroll
      for (int v = 0; v < 8; ++v) {
        int m = tileM + wm * 32 + i * 16 + half * 8 + v;
        int n = tileN + wn * 64 + j * 16 + l16;
        if (m < M && n < Nc) {
          float x = acc[i][j][v];
          if (bias) x += bias[n];
          if (act)  x = fmaxf(x, 0.0f);
          long long o = cofs + (long long)m * ldc + n;
          if (Cf) Cf[o] = x;
          if (Cb) Cb[o] = f2bf(x);
        }
      }
}

// g = sum_k graph[k]*wf[k]  (7x1024x1024 reduce), emitted as bf16.
__global__ void k_collapse(const float* __restrict__ g, const float* __restrict__ wf,
                           __bf16* __restrict__ out) {
  long long i = (long long)blockIdx.x * blockDim.x + threadIdx.x;
  const long long nn = (long long)NNODE * NNODE;
  if (i >= nn) return;
  float s = 0.f;
#pragma unroll
  for (int k = 0; k < GG; ++k) s += g[(long long)k * nn + i] * wf[(long long)k * nn + i];
  out[i] = f2bf(s);
}

// Strided f32 -> bf16 convert (slice/pack).
__global__ void k_cvt(const float* __restrict__ src, int ldS, int colOff,
                      __bf16* __restrict__ dst, int ldD, int dColOff,
                      long long rows, int cols) {
  long long i = (long long)blockIdx.x * blockDim.x + threadIdx.x;
  if (i >= rows * (long long)cols) return;
  long long r = i / cols; int c = (int)(i % cols);
  dst[r * ldD + dColOff + c] = f2bf(src[r * ldS + colOff + c]);
}

// Batched f32 -> bf16 transpose: dst[z][c*rows + r] = src[z][r*ldS + colOff + c].
__global__ void k_cvtT(const float* __restrict__ src, __bf16* __restrict__ dst,
                       int rows, int cols, int ldS, int colOff,
                       long long sSrc, long long sDst, int batch) {
  long long i = (long long)blockIdx.x * blockDim.x + threadIdx.x;
  long long per = (long long)rows * cols;
  if (i >= per * batch) return;
  int z = (int)(i / per); long long rm = i % per;
  int r = (int)(rm / cols), c = (int)(rm % cols);
  dst[(long long)z * sDst + (long long)c * rows + r] =
      f2bf(src[(long long)z * sSrc + (long long)r * ldS + colOff + c]);
}

__global__ void k_fill(float* p, long long n, float v) {
  long long i = (long long)blockIdx.x * blockDim.x + threadIdx.x;
  if (i < n) p[i] = v;
}
__global__ void k_deg(const int* __restrict__ dst, float* __restrict__ deg, long long e) {
  long long i = (long long)blockIdx.x * blockDim.x + threadIdx.x;
  if (i < e) atomicAdd(&deg[dst[i]], 1.0f);
}
__global__ void k_dinv(float* p, long long n) {  // deg>0 always (self-loop adds 1)
  long long i = (long long)blockIdx.x * blockDim.x + threadIdx.x;
  if (i < n) p[i] = rsqrtf(p[i] + 1.0f);
}

// out[i,c] = hw[i,c]*dinv[i]^2 + b[c]   (self-loop term + bias), cols = 128.
__global__ void k_gcn_init(const float* __restrict__ hw, const float* __restrict__ dinv,
                           const float* __restrict__ bias, float* __restrict__ out,
                           long long rows) {
  long long i = (long long)blockIdx.x * blockDim.x + threadIdx.x;
  if (i >= rows * 128) return;
  long long r = i >> 7; int c = (int)(i & 127);
  float di = dinv[r];
  out[i] = hw[i] * di * di + bias[c];
}

// out[dst] += hw[src] * dinv[src]*dinv[dst]; 4 channels/thread, cols = 128.
__global__ void k_gcn_scatter(const float* __restrict__ hw, const int* __restrict__ src,
                              const int* __restrict__ dst, const float* __restrict__ dinv,
                              float* __restrict__ out, long long E) {
  long long t = (long long)blockIdx.x * blockDim.x + threadIdx.x;
  if (t >= E * 32) return;
  long long e = t >> 5; int c0 = (int)((t & 31) << 2);
  int s = src[e], d = dst[e];
  float nrm = dinv[s] * dinv[d];
  const float4 v = *(const float4*)(hw + ((long long)s << 7) + c0);
  float* o = out + ((long long)d << 7) + c0;
  atomicAdd(o + 0, v.x * nrm); atomicAdd(o + 1, v.y * nrm);
  atomicAdd(o + 2, v.z * nrm); atomicAdd(o + 3, v.w * nrm);
}

// GLU (128 -> 64) + residual; optional f32 / bf16 outputs at ldo/colOff.
__global__ void k_glu(const float* __restrict__ hw, const float* __restrict__ res,
                      float* __restrict__ outF, __bf16* __restrict__ outB,
                      long long rows, int ldo, int colOff) {
  long long i = (long long)blockIdx.x * blockDim.x + threadIdx.x;
  if (i >= rows * 64) return;
  long long r = i >> 6; int c = (int)(i & 63);
  float a = hw[(r << 7) + c];
  float g = hw[(r << 7) + 64 + c];
  float y = a * (1.0f / (1.0f + __expf(-g)));
  if (res) y += res[(r << 6) + c];
  long long o = r * (long long)ldo + colOff + c;
  if (outF) outF[o] = y;
  if (outB) outB[o] = f2bf(y);
}

// BatchNorm stats: one workgroup per channel; biased var (ddof=0) like jnp.var.
__global__ __launch_bounds__(256) void k_bn_stats(const float* __restrict__ xin,
                                                  long long rows, int cols,
                                                  float* __restrict__ mean,
                                                  float* __restrict__ rstd) {
  __shared__ float ss[256], sq[256];
  int c = blockIdx.x, t = threadIdx.x;
  float s = 0.f, q = 0.f;
  for (long long r = t; r < rows; r += 256) {
    float v = xin[r * cols + c]; s += v; q += v * v;
  }
  ss[t] = s; sq[t] = q; __syncthreads();
  for (int st = 128; st > 0; st >>= 1) {
    if (t < st) { ss[t] += ss[t + st]; sq[t] += sq[t + st]; }
    __syncthreads();
  }
  if (t == 0) {
    float m = ss[0] / (float)rows;
    float v = sq[0] / (float)rows - m * m;
    mean[c] = m; rstd[c] = rsqrtf(v + 1e-5f);
  }
}

__global__ void k_bn_apply(const float* __restrict__ xin, const float* __restrict__ mean,
                           const float* __restrict__ rstd, const float* __restrict__ gamma,
                           const float* __restrict__ beta, __bf16* __restrict__ outB,
                           long long rows, int cols) {
  long long i = (long long)blockIdx.x * blockDim.x + threadIdx.x;
  if (i >= rows * cols) return;
  int c = (int)(i % cols);
  float y = (xin[i] - mean[c]) * rstd[c] * gamma[c] + beta[c];
  outB[i] = f2bf(fmaxf(y, 0.f));
}

// ---------------------------------------------------------------------------
extern "C" void kernel_launch(void* const* d_in, const int* in_sizes, int n_in,
                              void* d_out, int out_size, void* d_ws, size_t ws_size,
                              hipStream_t stream) {
  (void)n_in; (void)out_size; (void)ws_size;
  const float* x        = (const float*)d_in[0];
  const float* graphs[2] = {(const float*)d_in[1], (const float*)d_in[2]};
  const float* x_fc     = (const float*)d_in[3];
  const int*   eiS      = (const int*)d_in[4];
  const int*   eiL      = (const int*)d_in[5];
  const long long Es = in_sizes[4] / 2, El = in_sizes[5] / 2;
  const int *srcS = eiS, *dstS = eiS + Es;
  const int *srcL = eiL, *dstL = eiL + El;

  // params leaves assuming JAX pytree flattening (dict keys sorted):
  // fc{W0..3,b0..3,beta0..2,gamma0..2}=0..13, fusion{W,b}=14..15,
  // lt{(W,b)x3}=16..21, now=22, period=40, trend=58.
  // st_p layer0 (down): Wres,Wt,bres,bt, in{b,w,wf}, out{b,w,wf} (10 leaves)
  // st_p layer1:        Wt,bt, in{b,w,wf}, out{b,w,wf}          (8 leaves)
  auto P = [&](int i) { return (const float*)d_in[7 + i]; };

  // workspace bump allocator
  char* wsp = (char*)d_ws; size_t cur = 0;
  auto alloc = [&](size_t bytes) {
    void* p = wsp + cur; cur += (bytes + 255) & ~(size_t)255; return p;
  };
  float*  dinvS = (float*) alloc((size_t)NROWS * 4);
  float*  dinvL = (float*) alloc((size_t)NROWS * 4);
  __bf16* wT    = (__bf16*)alloc((size_t)NNODE * NNODE * 2);
  __bf16* gB    = (__bf16*)alloc((size_t)NNODE * NNODE * 2);
  __bf16* gwB   = (__bf16*)alloc((size_t)NNODE * NNODE * 2);
  __bf16* xT    = (__bf16*)alloc((size_t)NB * 64 * NNODE * 2);
  __bf16* x2d   = (__bf16*)alloc((size_t)NROWS * 64 * 2);
  float*  hcatF = (float*) alloc((size_t)NROWS * 128 * 4);
  __bf16* hcatB = (__bf16*)alloc((size_t)NROWS * 128 * 2);
  float*  hwF   = (float*) alloc((size_t)NROWS * 128 * 4);
  float*  gcnF  = (float*) alloc((size_t)NROWS * 128 * 4);
  float*  resF  = (float*) alloc((size_t)NROWS * 64 * 4);
  float*  out0F = (float*) alloc((size_t)NROWS * 64 * 4);
  __bf16* cat192= (__bf16*)alloc((size_t)NROWS * 192 * 2);
  float*  hFa   = (float*) alloc((size_t)NROWS * 64 * 4);
  float*  hFb   = (float*) alloc((size_t)NROWS * 64 * 4);
  __bf16* hBa   = (__bf16*)alloc((size_t)NROWS * 64 * 2);
  __bf16* hBb   = (__bf16*)alloc((size_t)NROWS * 64 * 2);
  __bf16* a72   = (__bf16*)alloc((size_t)NROWS * 72 * 2);
  float*  fcF   = (float*) alloc((size_t)NROWS * 64 * 4);
  __bf16* fcB   = (__bf16*)alloc((size_t)NROWS * 64 * 2);
  float*  bnM   = (float*) alloc(64 * 4);
  float*  bnR   = (float*) alloc(64 * 4);
  __bf16* wSmall= (__bf16*)alloc(64 * 1024);  // small weight transposes (<=32KB)

  auto L1 = [&](long long n) { return dim3((unsigned)((n + 255) / 256)); };
  auto gemm = [&](const __bf16* A, const __bf16* Bt, const float* bias,
                  float* Cf, __bf16* Cb, int M, int Nc, int K, int act, int ldc,
                  int batch, long long sA, long long sB, long long sC) {
    dim3 grid((Nc + BN - 1) / BN, (M + BM - 1) / BM, batch);
    k_gemm_bf16<<<grid, 256, 0, stream>>>(A, Bt, bias, Cf, Cb, M, Nc, K, act, ldc, sA, sB, sC);
  };
  auto cvt = [&](const float* s, int ldS, int co, __bf16* d, int ldD, int dco,
                 long long rows, int cols) {
    k_cvt<<<L1(rows * (long long)cols), 256, 0, stream>>>(s, ldS, co, d, ldD, dco, rows, cols);
  };
  auto cvtT = [&](const float* s, __bf16* d, int rows, int cols, int ldS, int co,
                  long long sS, long long sD, int batch) {
    k_cvtT<<<L1((long long)rows * cols * batch), 256, 0, stream>>>(s, d, rows, cols, ldS, co, sS, sD, batch);
  };

  // Degrees -> dinv (shared by all GCN calls).
  k_fill<<<L1(NROWS), 256, 0, stream>>>(dinvS, NROWS, 0.f);
  k_fill<<<L1(NROWS), 256, 0, stream>>>(dinvL, NROWS, 0.f);
  k_deg <<<L1(Es),    256, 0, stream>>>(dstS, dinvS, Es);
  k_deg <<<L1(El),    256, 0, stream>>>(dstL, dinvL, El);
  k_dinv<<<L1(NROWS), 256, 0, stream>>>(dinvS, NROWS);
  k_dinv<<<L1(NROWS), 256, 0, stream>>>(dinvL, NROWS);

  auto stconv = [&](int pb, bool down, int cin, const __bf16* xTb, const __bf16* x2db,
                    const float* resSrc, float* outF, __bf16* outB, int ldo, int co) {
    const int c2 = 2 * cin;
    const int iWt = down ? 1 : 0, ibt = down ? 3 : 1;
    const int mg[2] = { down ? 4 : 2, down ? 7 : 5 };   // in / out mgc leaf bases
    for (int br = 0; br < 2; ++br) {
      const float* mb  = P(pb + mg[br] + 0);
      const float* mw  = P(pb + mg[br] + 1);
      const float* mwf = P(pb + mg[br] + 2);
      cvtT(mw, wT, NNODE, NNODE, NNODE, 0, 0, 0, 1);                 // w -> w^T bf16
      k_collapse<<<L1((long long)NNODE * NNODE), 256, 0, stream>>>(graphs[br], mwf, gB);
      gemm(gB, wT, nullptr, nullptr, gwB, NNODE, NNODE, NNODE, 0, NNODE, 1, 0, 0, 0);
      // out[b] = (g@w) @ x[b] + bias, relu, packed into concat buffer.
      gemm(gwB, xTb, mb, hcatF + br * cin, hcatB + br * cin, NNODE, cin, NNODE, 1, c2,
           NB, 0, (long long)cin * NNODE, (long long)NNODE * c2);
    }
    // GCN: hw = h @ Wt
    cvtT(P(pb + iWt), wSmall, c2, 128, 128, 0, 0, 0, 1);
    gemm(hcatB, wSmall, nullptr, hwF, nullptr, NROWS, 128, c2, 0, 128, 1, 0, 0, 0);
    k_gcn_init<<<L1((long long)NROWS * 128), 256, 0, stream>>>(hwF, dinvS, P(pb + ibt), gcnF, NROWS);
    k_gcn_scatter<<<L1(Es * 32), 256, 0, stream>>>(hwF, srcS, dstS, dinvS, gcnF, Es);
    const float* res;
    if (down) {
      cvtT(P(pb + 0), wSmall, cin, 64, 64, 0, 0, 0, 1);              // Wres^T
      gemm(x2db, wSmall, P(pb + 2), resF, nullptr, NROWS, 64, cin, 1, 64, 1, 0, 0, 0);
      res = resF;
    } else {
      res = resSrc;
    }
    k_glu<<<L1((long long)NROWS * 64), 256, 0, stream>>>(gcnF, res, outF, outB, NROWS, ldo, co);
  };

  const int base[3] = {22, 40, 58};
  const int cin0[3] = {16, 32, 16};
  const int xOff[3] = {0, 16, 48};
  for (int s = 0; s < 3; ++s) {
    // layer 0 (down): x slice -> xT (B,C,N) and x2d (B*N,C) bf16
    cvtT(x, xT, NNODE, cin0[s], 64, xOff[s], (long long)NNODE * 64,
         (long long)cin0[s] * NNODE, NB);
    cvt (x, 64, xOff[s], x2d, cin0[s], 0, NROWS, cin0[s]);
    stconv(base[s], true, cin0[s], xT, x2d, nullptr, out0F, nullptr, 64, 0);
    // layer 1: residual = layer-0 output; result -> concat bf16 at col s*64
    cvtT(out0F, xT, NNODE, 64, 64, 0, (long long)NNODE * 64, (long long)64 * NNODE, NB);
    stconv(base[s] + 10, false, 64, xT, nullptr, out0F, nullptr, cat192, 192, s * 64);
  }

  // fusion: h = relu(cat192 @ Wfus + bfus)
  cvtT(P(14), wSmall, 192, 64, 64, 0, 0, 0, 1);
  gemm(cat192, wSmall, P(15), hFa, hBa, NROWS, 64, 192, 1, 64, 1, 0, 0, 0);

  float* hF = hFa; float* hF2 = hFb; __bf16* hB = hBa; __bf16* hB2 = hBb;
  for (int i = 0; i < 3; ++i) {   // lt layers: h = glu(gcn(h @ W)) + h
    cvtT(P(16 + 2 * i), wSmall, 64, 128, 128, 0, 0, 0, 1);
    gemm(hB, wSmall, nullptr, hwF, nullptr, NROWS, 128, 64, 0, 128, 1, 0, 0, 0);
    k_gcn_init<<<L1((long long)NROWS * 128), 256, 0, stream>>>(hwF, dinvL, P(17 + 2 * i), gcnF, NROWS);
    k_gcn_scatter<<<L1(El * 32), 256, 0, stream>>>(hwF, srcL, dstL, dinvL, gcnF, El);
    k_glu<<<L1((long long)NROWS * 64), 256, 0, stream>>>(gcnF, hF, hF2, hB2, NROWS, 64, 0);
    { float* tf = hF; hF = hF2; hF2 = tf; __bf16* tb = hB; hB = hB2; hB2 = tb; }
  }

  // fc head: concat(h, x_fc) -> 72 -> 64 -> 32 -> 16 -> 2, BN+relu between.
  cvt(hF,  64, 0, a72, 72, 0,  NROWS, 64);
  cvt(x_fc, 8, 0, a72, 72, 64, NROWS, 8);
  const int fdim[5] = {72, 64, 32, 16, 2};
  const __bf16* Ab = a72;
  for (int i = 0; i < 4; ++i) {
    int kd = fdim[i], nd = fdim[i + 1];
    cvtT(P(i), wSmall, kd, nd, nd, 0, 0, 0, 1);   // fc.W[i]^T bf16
    if (i < 3) {
      gemm(Ab, wSmall, P(4 + i), fcF, nullptr, NROWS, nd, kd, 0, nd, 1, 0, 0, 0);
      k_bn_stats<<<nd, 256, 0, stream>>>(fcF, NROWS, nd, bnM, bnR);
      k_bn_apply<<<L1((long long)NROWS * nd), 256, 0, stream>>>(
          fcF, bnM, bnR, P(11 + i), P(8 + i), fcB, NROWS, nd);  // gamma, beta
      Ab = fcB;
    } else {
      gemm(Ab, wSmall, P(4 + i), (float*)d_out, nullptr, NROWS, nd, kd, 1, nd, 1, 0, 0, 0);
    }
  }
}
